// DeepGRUPoolRecoveryHead_24721831756658
// MI455X (gfx1250) — matrix operand order repaired and checked by reading, not compiled, not run on hardware
//
#include <hip/hip_runtime.h>
#include <cstddef>

typedef unsigned short u16;
typedef unsigned int u32;
typedef __attribute__((ext_vector_type(16))) __bf16 v16bf;
typedef __attribute__((ext_vector_type(8))) float v8f;

struct Frag32 { uint4 a, b; };           // 32 bytes: one lane's 16 bf16 B-operand elems
struct AU8 { u32 d[8]; };                // 32 bytes: one lane's 16 bf16 A-operand elems

__device__ __forceinline__ u16 f2bf(float f) {
  u32 u = __float_as_uint(f);
  u32 r = u + 0x7FFFu + ((u >> 16) & 1u);  // round-to-nearest-even
  return (u16)(r >> 16);
}
// Branchless fast activations: v_exp_f32 + v_rcp_f32, no exec-mask piecewise code.
__device__ __forceinline__ float sigmoidf_(float x) {
  return __builtin_amdgcn_rcpf(1.0f + __expf(-x));
}
__device__ __forceinline__ float siluf_(float x) {
  return x * __builtin_amdgcn_rcpf(1.0f + __expf(-x));
}
__device__ __forceinline__ float tanhf_(float x) {
  return 1.0f - 2.0f * __builtin_amdgcn_rcpf(__expf(2.0f * x) + 1.0f);
}

// ---------------------------------------------------------------------------
// Pack a [N,K] fp32 weight (y = x @ W^T) into WMMA B-fragment order, bf16.
// Fragment f = nt*KT + kt covers B[k in kt*32..+31][n in nt*16..+15].
// Lane layout (v_wmma_*_16x16x32_bf16 B operand): h = lane/16; elem j<8 -> k = 8h + j,
// elem j>=8 -> k = 16 + 8h + (j-8) of the chunk; n = lane%16.
// ---------------------------------------------------------------------------
__global__ void pack_w(const float* __restrict__ w, u16* __restrict__ dst, int N, int K) {
  int KT = K >> 5;
  int f = blockIdx.x;
  int nt = f / KT, kt = f - nt * KT;
  int lane = threadIdx.x;
  int n = nt * 16 + (lane & 15);
  int kb = kt * 32 + ((lane >> 4) << 3);
  const float* src = w + (size_t)n * K + kb;
  u16* d = dst + ((size_t)f * 32 + lane) * 16;
#pragma unroll
  for (int j = 0; j < 16; ++j) d[j] = f2bf(src[j < 8 ? j : j + 8]);
}

// ---------------------------------------------------------------------------
// Tiled WMMA GEMM: C[M,N] = ACT(A[M,K](fp32) x packedW(bf16,[N,K]^T) + bias)
// Block 256 threads = 8 waves; 64x64 tile; wave (mt, half) does 16 rows x 32 cols.
// Double-buffered A tile in LDS -> single barrier per K-chunk.
// ACT template: 0=none, 1=SiLU, 2=tanh (straight-line epilogue, no branches).
// M%64==0, N%64==0, K%32==0.
// ---------------------------------------------------------------------------
template <int ACT>
__global__ __launch_bounds__(256) void gemm_bf16(
    const float* __restrict__ A, const u16* __restrict__ Bp,
    const float* __restrict__ bias, float* __restrict__ C,
    int M, int N, int K) {
  __shared__ u16 lds_a[2][64 * 40];  // 2 x (64 rows x 32 bf16, stride 40)
  const int tid = threadIdx.x;
  const int lane = tid & 31;
  const int wave = tid >> 5;
  const int mt = wave >> 1;
  const int half = wave & 1;
  const int bm = blockIdx.x;
  const int bn = blockIdx.y;
  const int KT = K >> 5;

  v8f acc0 = {0.f, 0.f, 0.f, 0.f, 0.f, 0.f, 0.f, 0.f};
  v8f acc1 = {0.f, 0.f, 0.f, 0.f, 0.f, 0.f, 0.f, 0.f};

  const int lr = tid >> 2;          // 0..63
  const int lc = (tid & 3) << 3;    // 0,8,16,24
  const int am = mt * 16 + (lane & 15);
  const int hi4 = (lane >> 4) << 2;

  for (int kt = 0; kt < KT; ++kt) {
    const int p = kt & 1;
    // Stage A tile: fp32 global -> bf16 LDS (8 elems/thread, b128 in/out)
    const float* ap = A + (size_t)(bm * 64 + lr) * K + kt * 32 + lc;
    float4 f0 = *(const float4*)ap;
    float4 f1 = *(const float4*)(ap + 4);
    uint4 q;
    q.x = (u32)f2bf(f0.x) | ((u32)f2bf(f0.y) << 16);
    q.y = (u32)f2bf(f0.z) | ((u32)f2bf(f0.w) << 16);
    q.z = (u32)f2bf(f1.x) | ((u32)f2bf(f1.y) << 16);
    q.w = (u32)f2bf(f1.z) | ((u32)f2bf(f1.w) << 16);
    *(uint4*)&lds_a[p][lr * 40 + lc] = q;
    __syncthreads();  // stores of chunk kt complete; buffer p readable

    // A fragment per ISA 16-bit A layout
    const u32* arow = (const u32*)&lds_a[p][am * 40];
    AU8 au;
#pragma unroll
    for (int pp = 0; pp < 8; ++pp) au.d[pp] = arow[(pp < 4 ? pp : pp + 4) + hi4];
    v16bf afrag = __builtin_bit_cast(v16bf, au);

#pragma unroll
    for (int ti = 0; ti < 2; ++ti) {
      int gnt = bn * 4 + half * 2 + ti;
      Frag32 braw = *(const Frag32*)(Bp + ((size_t)(gnt * KT + kt) * 32 + lane) * 16);
      v16bf bfrag = __builtin_bit_cast(v16bf, braw);
      if (ti == 0)
        acc0 = __builtin_amdgcn_wmma_f32_16x16x32_bf16(false, afrag, false, bfrag, (short)0, acc0, false, false);
      else
        acc1 = __builtin_amdgcn_wmma_f32_16x16x32_bf16(false, afrag, false, bfrag, (short)0, acc1, false, false);
    }
    // no second barrier: next iteration writes the other buffer, and its
    // barrier orders those stores after every wave's compute of chunk kt-1.
  }

  const int mrow0 = bm * 64 + mt * 16 + ((lane >> 4) << 3);
#pragma unroll
  for (int ti = 0; ti < 2; ++ti) {
    const int ncol = bn * 64 + (half * 2 + ti) * 16 + (lane & 15);
    const float bv = bias[ncol];
    v8f acc = (ti == 0) ? acc0 : acc1;
    float* crow = C + (size_t)mrow0 * N + ncol;
#pragma unroll
    for (int v = 0; v < 8; ++v) {
      float x = acc[v] + bv;
      if (ACT == 1) x = siluf_(x);
      if (ACT == 2) x = tanhf_(x);
      crow[(size_t)v * N] = x;
    }
  }
}

// ---------------------------------------------------------------------------
// Row-wise LayerNorm, in place. One 256-thread block per row.
// ---------------------------------------------------------------------------
__global__ __launch_bounds__(256) void ln_kernel(float* __restrict__ x,
                                                 const float* __restrict__ g,
                                                 const float* __restrict__ b, int D) {
  __shared__ float red[256];
  const int tid = threadIdx.x;
  const float rD = 1.0f / (float)D;
  float* xr = x + (size_t)blockIdx.x * D;
  float s = 0.f;
  for (int j = tid; j < D; j += 256) s += xr[j];
  red[tid] = s; __syncthreads();
  for (int st = 128; st > 0; st >>= 1) { if (tid < st) red[tid] += red[tid + st]; __syncthreads(); }
  float mean = red[0] * rD; __syncthreads();
  float v = 0.f;
  for (int j = tid; j < D; j += 256) { float d = xr[j] - mean; v += d * d; }
  red[tid] = v; __syncthreads();
  for (int st = 128; st > 0; st >>= 1) { if (tid < st) red[tid] += red[tid + st]; __syncthreads(); }
  float inv = rsqrtf(red[0] * rD + 1e-5f);
  for (int j = tid; j < D; j += 256) xr[j] = (xr[j] - mean) * inv * g[j] + b[j];
}

// ---------------------------------------------------------------------------
// Persistent GRU recurrence. grid=2 (block 0: fwd, block 1: bwd), 1024 thr.
// Per step: hg = h @ w_hh^T via WMMA (192 tiles / 32 waves), then gates.
// LDS: h fp32 (64KB) + h bf16 copy (33KB) + hg (192KB) + b_hh (3KB) = 292KB.
// An asm memory clobber per step stops LICM from hoisting the 48 loop-
// invariant B-fragment loads (384 VGPRs -> scratch spill); they re-stream
// from L2 each step, overlapped with the WMMA chain.
// ---------------------------------------------------------------------------
__global__ __launch_bounds__(1024) void gru_rec(
    const float* __restrict__ xgf, const float* __restrict__ xgb,
    const u16* __restrict__ whf, const u16* __restrict__ whb,
    const float* __restrict__ bhf, const float* __restrict__ bhb,
    float* __restrict__ out) {
  extern __shared__ char smem[];
  float* h32 = (float*)smem;                                  // 64*256 fp32
  u16* hbf = (u16*)(smem + 64 * 256 * 4);                     // 64 rows, stride 264 u16
  float* hg = (float*)(smem + 64 * 256 * 4 + 64 * 264 * 2);   // 64*768 fp32
  float* bhs = (float*)(smem + 64 * 256 * 4 + 64 * 264 * 2 + 64 * 768 * 4);  // 768 fp32

  const int tid = threadIdx.x;
  const int lane = tid & 31;
  const int wave = tid >> 5;
  const int dir = blockIdx.x;
  const float* xg = dir ? xgb : xgf;
  const u16* wp = dir ? whb : whf;
  const float* bhh = dir ? bhb : bhf;
  const int col0 = dir ? 256 : 0;

  const int mt = wave >> 3;          // 0..3 (row-tile of h)
  const int nbase = (wave & 7) * 6;  // 6 N-tiles per wave, 48 total

  for (int i = tid; i < 64 * 256; i += 1024) h32[i] = 0.0f;
  for (int i = tid; i < 64 * 264; i += 1024) hbf[i] = 0;
  if (tid < 768) bhs[tid] = bhh[tid];
  __syncthreads();

  const int am = mt * 16 + (lane & 15);
  const int hi4 = (lane >> 4) << 2;
  const int brow = mt * 16 + ((lane >> 4) << 3);
  const int ncl = lane & 15;

  for (int t = 0; t < 512; ++t) {
    const int tc = dir ? (511 - t) : t;

    // Alias barrier: forces the weight-fragment loads to stay inside the
    // loop (re-streamed from L2) instead of being hoisted + spilled.
    asm volatile("" ::: "memory");

    // Prefetch a slice of this step's xg gate data behind the WMMA phase.
#pragma unroll
    for (int e2 = 0; e2 < 16; e2 += 4) {
      int idx = e2 * 1024 + tid;
      int b = idx >> 8, j = idx & 255;
      const float* xp = xg + ((size_t)b * 512 + tc) * 768 + j;
      __builtin_prefetch(xp, 0, 3);
      __builtin_prefetch(xp + 256, 0, 3);
      __builtin_prefetch(xp + 512, 0, 3);
    }

    v8f acc[6];
#pragma unroll
    for (int i = 0; i < 6; ++i) acc[i] = (v8f){0.f, 0.f, 0.f, 0.f, 0.f, 0.f, 0.f, 0.f};

#pragma unroll
    for (int kt = 0; kt < 8; ++kt) {
      const u32* arow = (const u32*)&hbf[am * 264 + kt * 32];
      AU8 au;
#pragma unroll
      for (int p = 0; p < 8; ++p) au.d[p] = arow[(p < 4 ? p : p + 4) + hi4];
      v16bf afrag = __builtin_bit_cast(v16bf, au);
#pragma unroll
      for (int i = 0; i < 6; ++i) {
        Frag32 braw = *(const Frag32*)(wp + ((size_t)((nbase + i) * 8 + kt) * 32 + lane) * 16);
        v16bf bfrag = __builtin_bit_cast(v16bf, braw);
        acc[i] = __builtin_amdgcn_wmma_f32_16x16x32_bf16(false, afrag, false, bfrag, (short)0, acc[i], false, false);
      }
    }
#pragma unroll
    for (int i = 0; i < 6; ++i) {
      int nc = (nbase + i) * 16 + ncl;
#pragma unroll
      for (int v = 0; v < 8; ++v) hg[(brow + v) * 768 + nc] = acc[i][v];
    }
    __syncthreads();

    // Gates: 64*256 = 16384 elements, 16 per thread
#pragma unroll
    for (int e = 0; e < 16; ++e) {
      int idx = e * 1024 + tid;
      int b = idx >> 8, j = idx & 255;
      size_t xrow = ((size_t)b * 512 + tc) * 768;
      float xr = xg[xrow + j];
      float xz = xg[xrow + 256 + j];
      float xn = xg[xrow + 512 + j];
      float hr = hg[b * 768 + j] + bhs[j];
      float hz = hg[b * 768 + 256 + j] + bhs[256 + j];
      float hn = hg[b * 768 + 512 + j] + bhs[512 + j];
      float r = sigmoidf_(xr + hr);
      float z = sigmoidf_(xz + hz);
      float n = tanhf_(xn + r * hn);
      float hold = h32[b * 256 + j];
      float hnew = (1.0f - z) * n + z * hold;
      h32[b * 256 + j] = hnew;
      hbf[b * 264 + j] = f2bf(hnew);
      out[((size_t)b * 512 + tc) * 512 + col0 + j] = hnew;
    }
    __syncthreads();
  }
}

// scores[r] = s[r,:] . a_w2 + a_b2   (r = 0..32767, dot over 256)
__global__ __launch_bounds__(256) void attn_score(const float* __restrict__ s,
                                                  const float* __restrict__ w2,
                                                  const float* __restrict__ b2,
                                                  float* __restrict__ sc) {
  int r = blockIdx.x * 256 + threadIdx.x;
  const float* row = s + (size_t)r * 256;
  float a = 0.f;
  for (int k = 0; k < 256; ++k) a += row[k] * w2[k];
  sc[r] = a + b2[0];
}

// softmax over T + weighted pooling; one block per batch element
__global__ __launch_bounds__(256) void attn_pool(const float* __restrict__ go,
                                                 const float* __restrict__ sc,
                                                 float* __restrict__ pooled) {
  __shared__ float w[512];
  __shared__ float red[256];
  const int b = blockIdx.x, tid = threadIdx.x;
  float m = -3.0e38f;
  for (int t = tid; t < 512; t += 256) { float v = sc[b * 512 + t]; w[t] = v; m = fmaxf(m, v); }
  red[tid] = m; __syncthreads();
  for (int st = 128; st > 0; st >>= 1) { if (tid < st) red[tid] = fmaxf(red[tid], red[tid + st]); __syncthreads(); }
  float mx = red[0]; __syncthreads();
  float s = 0.f;
  for (int t = tid; t < 512; t += 256) { float e = __expf(w[t] - mx); w[t] = e; s += e; }
  red[tid] = s; __syncthreads();
  for (int st = 128; st > 0; st >>= 1) { if (tid < st) red[tid] += red[tid + st]; __syncthreads(); }
  float inv = __builtin_amdgcn_rcpf(red[0]); __syncthreads();
  for (int t = tid; t < 512; t += 256) w[t] *= inv;
  __syncthreads();
  for (int c = tid; c < 512; c += 256) {
    float a = 0.f;
    for (int t = 0; t < 512; ++t) a += go[((size_t)b * 512 + t) * 512 + c] * w[t];
    pooled[b * 512 + c] = a;
  }
}

// output MLP + heads; one block per batch element. out: [ar(64*4) | ma(64*4)]
__global__ __launch_bounds__(256) void head_kernel(
    const float* __restrict__ pooled,
    const float* __restrict__ w1, const float* __restrict__ b1,
    const float* __restrict__ g1, const float* __restrict__ be1,
    const float* __restrict__ w2, const float* __restrict__ b2,
    const float* __restrict__ w3, const float* __restrict__ b3,
    const float* __restrict__ arw, const float* __restrict__ arb,
    const float* __restrict__ maw, const float* __restrict__ mab,
    float* __restrict__ out) {
  __shared__ float p[512], f1[256], f2[256], f3[256], red[256];
  const int b = blockIdx.x, tid = threadIdx.x;
  for (int j = tid; j < 512; j += 256) p[j] = pooled[b * 512 + j];
  __syncthreads();
  float a = 0.f;
  const float* wr = w1 + (size_t)tid * 512;
  for (int k = 0; k < 512; ++k) a += p[k] * wr[k];
  a = siluf_(a + b1[tid]);
  red[tid] = a; __syncthreads();
  for (int st = 128; st > 0; st >>= 1) { if (tid < st) red[tid] += red[tid + st]; __syncthreads(); }
  float mean = red[0] * (1.0f / 256.0f); __syncthreads();
  float d = a - mean;
  red[tid] = d * d; __syncthreads();
  for (int st = 128; st > 0; st >>= 1) { if (tid < st) red[tid] += red[tid + st]; __syncthreads(); }
  float inv = rsqrtf(red[0] * (1.0f / 256.0f) + 1e-5f);
  f1[tid] = d * inv * g1[tid] + be1[tid];
  __syncthreads();
  float a2 = 0.f;
  const float* wr2 = w2 + (size_t)tid * 256;
  for (int k = 0; k < 256; ++k) a2 += f1[k] * wr2[k];
  f2[tid] = siluf_(a2 + b2[tid]);
  __syncthreads();
  float a3 = 0.f;
  const float* wr3 = w3 + (size_t)tid * 256;
  for (int k = 0; k < 256; ++k) a3 += f2[k] * wr3[k];
  f3[tid] = siluf_(a3 + b3[tid]);
  __syncthreads();
  if (tid < 4) {
    float sacc = 0.f;
    const float* w = arw + tid * 256;
    for (int k = 0; k < 256; ++k) sacc += f3[k] * w[k];
    out[b * 4 + tid] = tanhf_(sacc + arb[tid]);
  } else if (tid < 8) {
    int q = tid - 4;
    float sacc = 0.f;
    const float* w = maw + q * 256;
    for (int k = 0; k < 256; ++k) sacc += f3[k] * w[k];
    out[256 + b * 4 + q] = tanhf_(sacc + mab[q]);
  }
}

// ---------------------------------------------------------------------------
extern "C" void kernel_launch(void* const* d_in, const int* in_sizes, int n_in,
                              void* d_out, int out_size, void* d_ws, size_t ws_size,
                              hipStream_t stream) {
  (void)in_sizes; (void)n_in; (void)out_size; (void)ws_size;
  const int M = 64 * 512;  // BC*T

  const float* IN[49];
  for (int i = 0; i < 49; ++i) IN[i] = (const float*)d_in[i];
  const float* x = IN[0];
  const float *ip_w1 = IN[1], *ip_b1 = IN[2], *ip_g1 = IN[3], *ip_be1 = IN[4];
  const float *ip_w2 = IN[5], *ip_b2 = IN[6], *ip_g2 = IN[7], *ip_be2 = IN[8];
  // GRU params: base 9, 8 per layer (f: wih,whh,bih,bhh; b: wih,whh,bih,bhh)
  const float *a_w1 = IN[33], *a_b1 = IN[34], *a_w2 = IN[35], *a_b2 = IN[36];
  const float *o_w1 = IN[37], *o_b1 = IN[38], *o_g1 = IN[39], *o_be1 = IN[40];
  const float *o_w2 = IN[41], *o_b2 = IN[42], *o_w3 = IN[43], *o_b3 = IN[44];
  const float *ar_w = IN[45], *ar_b = IN[46], *ma_w = IN[47], *ma_b = IN[48];

  // ---- workspace carve ----
  char* ws = (char*)d_ws;
  size_t off = 0;
  auto alloc = [&](size_t bytes) -> char* {
    char* p = ws + off;
    off += (bytes + 255) & ~(size_t)255;
    return p;
  };
  u16* pk_ipw1 = (u16*)alloc((size_t)512 * 1024 * 2);
  u16* pk_ipw2 = (u16*)alloc((size_t)256 * 512 * 2);
  const int din[3] = {256, 512, 512};
  u16 *pk_wih[6], *pk_whh[6];
  for (int l = 0; l < 3; ++l)
    for (int d = 0; d < 2; ++d) {
      pk_wih[l * 2 + d] = (u16*)alloc((size_t)768 * din[l] * 2);
      pk_whh[l * 2 + d] = (u16*)alloc((size_t)768 * 256 * 2);
    }
  u16* pk_aw1 = (u16*)alloc((size_t)256 * 512 * 2);
  float* H1 = (float*)alloc((size_t)M * 512 * 4);
  float* H2 = (float*)alloc((size_t)M * 256 * 4);
  float* XGF = (float*)alloc((size_t)M * 768 * 4);
  float* XGB = (float*)alloc((size_t)M * 768 * 4);
  float* GA = (float*)alloc((size_t)M * 512 * 4);
  float* GB = (float*)alloc((size_t)M * 512 * 4);
  float* SCORES = (float*)alloc((size_t)M * 4);
  float* POOLED = (float*)alloc((size_t)64 * 512 * 4);

  // ---- pack weights to bf16 fragments ----
  pack_w<<<(512 / 16) * (1024 / 32), 32, 0, stream>>>(ip_w1, pk_ipw1, 512, 1024);
  pack_w<<<(256 / 16) * (512 / 32), 32, 0, stream>>>(ip_w2, pk_ipw2, 256, 512);
  for (int l = 0; l < 3; ++l)
    for (int d = 0; d < 2; ++d) {
      const float* wih = IN[9 + l * 8 + d * 4 + 0];
      const float* whh = IN[9 + l * 8 + d * 4 + 1];
      pack_w<<<(768 / 16) * (din[l] / 32), 32, 0, stream>>>(wih, pk_wih[l * 2 + d], 768, din[l]);
      pack_w<<<(768 / 16) * (256 / 32), 32, 0, stream>>>(whh, pk_whh[l * 2 + d], 768, 256);
    }
  pack_w<<<(256 / 16) * (512 / 32), 32, 0, stream>>>(a_w1, pk_aw1, 256, 512);

  // ---- input projection ----
  gemm_bf16<1><<<dim3(M / 64, 512 / 64), 256, 0, stream>>>(x, pk_ipw1, ip_b1, H1, M, 512, 1024);
  ln_kernel<<<M, 256, 0, stream>>>(H1, ip_g1, ip_be1, 512);
  gemm_bf16<1><<<dim3(M / 64, 256 / 64), 256, 0, stream>>>(H1, pk_ipw2, ip_b2, H2, M, 256, 512);
  ln_kernel<<<M, 256, 0, stream>>>(H2, ip_g2, ip_be2, 256);

  // ---- 3-layer bidirectional GRU ----
  const size_t REC_LDS = (size_t)64 * 256 * 4 + (size_t)64 * 264 * 2 + (size_t)64 * 768 * 4 + 768 * 4;
  hipFuncSetAttribute((const void*)gru_rec, hipFuncAttributeMaxDynamicSharedMemorySize, (int)REC_LDS);

  const float* lin = H2;
  int lk = 256;
  float* louts[3] = {GA, GB, GA};
  for (int l = 0; l < 3; ++l) {
    const float* bih_f = IN[9 + l * 8 + 2];
    const float* bhh_f = IN[9 + l * 8 + 3];
    const float* bih_b = IN[9 + l * 8 + 6];
    const float* bhh_b = IN[9 + l * 8 + 7];
    gemm_bf16<0><<<dim3(M / 64, 768 / 64), 256, 0, stream>>>(lin, pk_wih[l * 2 + 0], bih_f, XGF, M, 768, lk);
    gemm_bf16<0><<<dim3(M / 64, 768 / 64), 256, 0, stream>>>(lin, pk_wih[l * 2 + 1], bih_b, XGB, M, 768, lk);
    gru_rec<<<2, 1024, REC_LDS, stream>>>(XGF, XGB, pk_whh[l * 2 + 0], pk_whh[l * 2 + 1],
                                          bhh_f, bhh_b, louts[l]);
    lin = louts[l];
    lk = 512;
  }
  float* GOUT = GA;  // final bidirectional output [M, 512]

  // ---- attention pooling ----
  gemm_bf16<2><<<dim3(M / 64, 256 / 64), 256, 0, stream>>>(GOUT, pk_aw1, a_b1, H1, M, 256, 512);
  attn_score<<<M / 256, 256, 0, stream>>>(H1, a_w2, a_b2, SCORES);
  attn_pool<<<64, 256, 0, stream>>>(GOUT, SCORES, POOLED);

  // ---- output MLP + heads ----
  head_kernel<<<64, 256, 0, stream>>>(POOLED, o_w1, o_b1, o_g1, o_be1, o_w2, o_b2,
                                      o_w3, o_b3, ar_w, ar_b, ma_w, ma_b, (float*)d_out);
}
